// SelfAttention_48352741819028
// MI455X (gfx1250) — matrix operand order repaired and checked by reading, not compiled, not run on hardware
//
#include <hip/hip_runtime.h>
#include <stdint.h>

#define SS 2048
#define BB 2
#define HH 1024
#define NHEAD 16
#define HDIM 64
#define MM (SS * BB) // 4096

typedef __attribute__((ext_vector_type(16))) __bf16 v16bf;
typedef __attribute__((ext_vector_type(8)))  float  v8f;

union Frag16 { v16bf v; unsigned int u[8]; unsigned short h[16]; };

__device__ __forceinline__ unsigned short f2bf(float f) {
  unsigned int x = __builtin_bit_cast(unsigned int, f);
  x += 0x7FFFu + ((x >> 16) & 1u);          // round-to-nearest-even
  return (unsigned short)(x >> 16);
}
__device__ __forceinline__ unsigned int pack2bf(float lo, float hi) {
  return (unsigned int)f2bf(lo) | ((unsigned int)f2bf(hi) << 16);
}

// A-matrix 16x32 bf16 fragment: first K of the packed pair in VGPR v for lane-group g
__device__ __forceinline__ int a_k0(int v, int g) {
  return ((v < 4) ? (2 * v) : (8 + 2 * v)) + 8 * g;
}
// B-matrix 32x16 bf16 fragment: first K of the packed pair in VGPR v for lane-group g
__device__ __forceinline__ int b_k0(int v, int g) {
  return ((v < 4) ? (2 * v) : (8 + 2 * v)) + 8 * g;
}

// ---------------------------------------------------------------------------
// Kernel 1: QKV projection. C = (X @ W^T + bias) * scale, scattered into
// split-head layout [B*NH, S, HD] as bf16.  grid = (H/64, M/64, 3)
// ---------------------------------------------------------------------------
__global__ __launch_bounds__(256)
void qkv_gemm(const float* __restrict__ X,
              const float* __restrict__ Wq, const float* __restrict__ bq,
              const float* __restrict__ Wk, const float* __restrict__ bk,
              const float* __restrict__ Wv, const float* __restrict__ bv,
              unsigned short* __restrict__ qb,
              unsigned short* __restrict__ kb,
              unsigned short* __restrict__ vb) {
  __shared__ unsigned short As[64][32];
  __shared__ unsigned short Bs[64][32];

  const int z = blockIdx.z;
  const float* W    = (z == 0) ? Wq : (z == 1) ? Wk : Wv;
  const float* bias = (z == 0) ? bq : (z == 1) ? bk : bv;
  unsigned short* outb = (z == 0) ? qb : (z == 1) ? kb : vb;
  const float scale = (z == 0) ? 0.125f : 1.0f;   // 1/sqrt(64) on Q only

  const int n0 = blockIdx.x * 64;
  const int m0 = blockIdx.y * 64;
  const int tid  = threadIdx.x;
  const int lane = tid & 31;
  const int wid  = tid >> 5;
  const int wm = wid & 3;     // 4 wave-slabs along M (16 rows each)
  const int wn = wid >> 2;    // 2 wave-slabs along N (32 cols each)
  const int g  = lane >> 4;
  const int ln = lane & 15;
  const int lr = tid >> 2;          // LDS fill row 0..63
  const int lc = (tid & 3) * 8;     // LDS fill col start

  v8f acc0 = {}; v8f acc1 = {};

  for (int k0 = 0; k0 < HH; k0 += 32) {
    { // A tile: X[m0+lr][k0+lc..+8), f32 -> bf16 packed pairs
      const float* src = X + (size_t)(m0 + lr) * HH + k0 + lc;
      unsigned int* dst = (unsigned int*)&As[lr][lc];
      #pragma unroll
      for (int i = 0; i < 4; ++i) dst[i] = pack2bf(src[2 * i], src[2 * i + 1]);
    }
    { // B tile: Bs[n][k] = W[n0+n][k0+k]
      const float* src = W + (size_t)(n0 + lr) * HH + k0 + lc;
      unsigned int* dst = (unsigned int*)&Bs[lr][lc];
      #pragma unroll
      for (int i = 0; i < 4; ++i) dst[i] = pack2bf(src[2 * i], src[2 * i + 1]);
    }
    __syncthreads();

    Frag16 a;
    #pragma unroll
    for (int v = 0; v < 8; ++v)
      a.u[v] = *(const unsigned int*)&As[wm * 16 + ln][a_k0(v, g)];

    Frag16 b0, b1;
    #pragma unroll
    for (int v = 0; v < 8; ++v) {
      b0.u[v] = *(const unsigned int*)&Bs[wn * 32 + ln][b_k0(v, g)];
      b1.u[v] = *(const unsigned int*)&Bs[wn * 32 + 16 + ln][b_k0(v, g)];
    }
    acc0 = __builtin_amdgcn_wmma_f32_16x16x32_bf16(false, a.v, false, b0.v,
                                                   (short)0, acc0, false, false);
    acc1 = __builtin_amdgcn_wmma_f32_16x16x32_bf16(false, a.v, false, b1.v,
                                                   (short)0, acc1, false, false);
    __syncthreads();
  }

  // Epilogue: bias, scale, scatter to [b*NH+nh][s][hd]
  #pragma unroll
  for (int nt = 0; nt < 2; ++nt) {
    v8f acc = nt ? acc1 : acc0;
    const int j  = n0 + wn * 32 + nt * 16 + ln;
    const float bj = bias[j];
    const int nh = j >> 6, hd = j & 63;
    #pragma unroll
    for (int v = 0; v < 8; ++v) {
      int m = m0 + wm * 16 + g * 8 + v;   // row in [S*B] (= s*B + b)
      int s = m >> 1, b = m & 1;
      int head = b * NHEAD + nh;
      float val = (acc[v] + bj) * scale;
      outb[((size_t)head * SS + s) * HDIM + hd] = f2bf(val);
    }
  }
}

// ---------------------------------------------------------------------------
// Kernel 2: flash attention. grid = (S/64, 32).  128 threads = 4 waves;
// each wave owns 16 query rows; K/V blocks of 64 keys staged in LDS.
// ---------------------------------------------------------------------------
__global__ __launch_bounds__(128)
void flash_attn(const unsigned short* __restrict__ qb,
                const unsigned short* __restrict__ kb,
                const unsigned short* __restrict__ vb,
                const float* __restrict__ amask,   // [NH, S, S]
                const float* __restrict__ kmask,   // [B*NH, S]
                unsigned short* __restrict__ ctx)  // [M, H] bf16
{
  __shared__ unsigned short Ks[64][64];       // [t][d]
  __shared__ unsigned short Vt[64][64];       // transposed: [d][t]
  __shared__ unsigned short Ps[4][16][64];    // per-wave P tile [s][t]

  const int n    = blockIdx.y;        // head 0..31
  const int nh   = n & 15;
  const int bidx = n >> 4;
  const int m0   = blockIdx.x * 64;   // query-row base (s)
  const int tid  = threadIdx.x;
  const int lane = tid & 31;
  const int w    = tid >> 5;
  const int g    = lane >> 4;
  const int ln   = lane & 15;

  // Q fragments for this wave's 16 rows, resident for the whole loop
  Frag16 qf[2];
  {
    const unsigned short* qrow =
        qb + ((size_t)n * SS + (m0 + w * 16 + ln)) * HDIM;
    #pragma unroll
    for (int c = 0; c < 2; ++c)
      #pragma unroll
      for (int v = 0; v < 8; ++v)
        qf[c].u[v] = *(const unsigned int*)&qrow[a_k0(v, g) + 32 * c];
  }

  float mrun[8], lrun[8];
  #pragma unroll
  for (int v = 0; v < 8; ++v) { mrun[v] = -1e30f; lrun[v] = 0.0f; }
  v8f zero = {};
  v8f oacc[4];
  #pragma unroll
  for (int dt = 0; dt < 4; ++dt) oacc[dt] = zero;

  for (int t0 = 0; t0 < SS; t0 += 64) {
    { // cooperative K block load (straight copy) + V block load (transpose)
      const uint4* ksrc = (const uint4*)(kb + ((size_t)n * SS + t0) * HDIM);
      uint4* kdst = (uint4*)&Ks[0][0];
      for (int i = tid; i < 512; i += 128) {
        kdst[i] = ksrc[i];
        if (t0 + 64 < SS) __builtin_prefetch(ksrc + 512 + i, 0, 0);
      }
      const uint4* vsrc = (const uint4*)(vb + ((size_t)n * SS + t0) * HDIM);
      for (int i = tid; i < 512; i += 128) {
        uint4 x = vsrc[i];
        if (t0 + 64 < SS) __builtin_prefetch(vsrc + 512 + i, 0, 0);
        int e = i * 8, t = e >> 6, d0 = e & 63;
        const unsigned short* hp = (const unsigned short*)&x;
        #pragma unroll
        for (int j2 = 0; j2 < 8; ++j2) Vt[d0 + j2][t] = hp[j2];
      }
    }
    __syncthreads();

    // scores = Q . K^T   (B fragment read from Ks: B[k=d][n=t])
    v8f sc[4];
    #pragma unroll
    for (int nt = 0; nt < 4; ++nt) {
      v8f a = zero;
      #pragma unroll
      for (int c = 0; c < 2; ++c) {
        Frag16 bfrag;
        #pragma unroll
        for (int v = 0; v < 8; ++v)
          bfrag.u[v] =
              *(const unsigned int*)&Ks[nt * 16 + ln][b_k0(v, g) + 32 * c];
        a = __builtin_amdgcn_wmma_f32_16x16x32_bf16(false, qf[c].v, false,
                                                    bfrag.v, (short)0, a,
                                                    false, false);
      }
      sc[nt] = a;
    }

    // add masks in f32 (C layout: s = srow+v, t = t0 + nt*16 + ln)
    const int srow = m0 + w * 16 + g * 8;
    #pragma unroll
    for (int nt = 0; nt < 4; ++nt) {
      int t = t0 + nt * 16 + ln;
      float km = kmask[(size_t)n * SS + t];
      const float* am = amask + ((size_t)nh * SS) * SS + t;
      #pragma unroll
      for (int v = 0; v < 8; ++v) {
        sc[nt][v] = sc[nt][v] + am[(size_t)(srow + v) * SS] + km;
        // prefetch next key-block's mask row (streams 256MB from HBM)
        if (t0 + 64 < SS)
          __builtin_prefetch(am + (size_t)(srow + v) * SS + 64, 0, 0);
      }
    }

    // online softmax; each (v, lane-group) is one row across 16 lanes x 4 tiles
    #pragma unroll
    for (int v = 0; v < 8; ++v) {
      float x = fmaxf(fmaxf(sc[0][v], sc[1][v]), fmaxf(sc[2][v], sc[3][v]));
      #pragma unroll
      for (int off = 1; off < 16; off <<= 1)
        x = fmaxf(x, __shfl_xor(x, off, 32));
      float mnew = fmaxf(mrun[v], x);
      float corr = __expf(mrun[v] - mnew);
      mrun[v] = mnew;
      float psum = 0.0f;
      #pragma unroll
      for (int nt = 0; nt < 4; ++nt) {
        float p = __expf(sc[nt][v] - mnew);
        sc[nt][v] = p;
        psum += p;
      }
      #pragma unroll
      for (int off = 1; off < 16; off <<= 1)
        psum += __shfl_xor(psum, off, 32);
      lrun[v] = lrun[v] * corr + psum;
      #pragma unroll
      for (int dt = 0; dt < 4; ++dt) oacc[dt][v] = oacc[dt][v] * corr;
    }

    // C-layout -> A-layout for P via per-wave LDS region (bf16)
    #pragma unroll
    for (int nt = 0; nt < 4; ++nt)
      #pragma unroll
      for (int v = 0; v < 8; ++v)
        Ps[w][v + 8 * g][nt * 16 + ln] = f2bf(sc[nt][v]);

    // O += P . V   (B fragment read from Vt: B[k=t][n=d])
    #pragma unroll
    for (int c = 0; c < 2; ++c) {
      Frag16 pf;
      #pragma unroll
      for (int v = 0; v < 8; ++v)
        pf.u[v] = *(const unsigned int*)&Ps[w][ln][a_k0(v, g) + 32 * c];
      #pragma unroll
      for (int dt = 0; dt < 4; ++dt) {
        Frag16 vf;
        #pragma unroll
        for (int v = 0; v < 8; ++v)
          vf.u[v] =
              *(const unsigned int*)&Vt[dt * 16 + ln][b_k0(v, g) + 32 * c];
        oacc[dt] = __builtin_amdgcn_wmma_f32_16x16x32_bf16(
            false, pf.v, false, vf.v, (short)0, oacc[dt], false, false);
      }
    }
    __syncthreads();
  }

  // normalize and store ctx bf16 at [s*B + b][nh*64 + d]
  #pragma unroll
  for (int v = 0; v < 8; ++v) {
    float inv = 1.0f / lrun[v];
    int s = m0 + w * 16 + g * 8 + v;
    size_t row = (size_t)s * BB + bidx;
    #pragma unroll
    for (int dt = 0; dt < 4; ++dt) {
      int d = dt * 16 + ln;
      ctx[row * HH + nh * HDIM + d] = f2bf(oacc[dt][v] * inv);
    }
  }
}

// ---------------------------------------------------------------------------
// Kernel 3: output projection.  out = ctx @ Wo^T + bo  (f32 out, [S,B,H] flat)
// grid = (H/64, M/64)
// ---------------------------------------------------------------------------
__global__ __launch_bounds__(256)
void out_proj(const unsigned short* __restrict__ A,  // ctx bf16 [M,H]
              const float* __restrict__ Wo, const float* __restrict__ bo,
              float* __restrict__ out) {
  __shared__ unsigned short As[64][32];
  __shared__ unsigned short Bs[64][32];
  const int n0 = blockIdx.x * 64;
  const int m0 = blockIdx.y * 64;
  const int tid  = threadIdx.x;
  const int lane = tid & 31;
  const int wid  = tid >> 5;
  const int wm = wid & 3, wn = wid >> 2;
  const int g = lane >> 4, ln = lane & 15;
  const int lr = tid >> 2, lc = (tid & 3) * 8;

  v8f acc0 = {}; v8f acc1 = {};
  for (int k0 = 0; k0 < HH; k0 += 32) {
    { // A already bf16: straight 16-byte copy
      const uint4* src = (const uint4*)(A + (size_t)(m0 + lr) * HH + k0 + lc);
      *(uint4*)&As[lr][lc] = *src;
    }
    {
      const float* src = Wo + (size_t)(n0 + lr) * HH + k0 + lc;
      unsigned int* dst = (unsigned int*)&Bs[lr][lc];
      #pragma unroll
      for (int i = 0; i < 4; ++i) dst[i] = pack2bf(src[2 * i], src[2 * i + 1]);
    }
    __syncthreads();
    Frag16 a;
    #pragma unroll
    for (int v = 0; v < 8; ++v)
      a.u[v] = *(const unsigned int*)&As[wm * 16 + ln][a_k0(v, g)];
    Frag16 b0, b1;
    #pragma unroll
    for (int v = 0; v < 8; ++v) {
      b0.u[v] = *(const unsigned int*)&Bs[wn * 32 + ln][b_k0(v, g)];
      b1.u[v] = *(const unsigned int*)&Bs[wn * 32 + 16 + ln][b_k0(v, g)];
    }
    acc0 = __builtin_amdgcn_wmma_f32_16x16x32_bf16(false, a.v, false, b0.v,
                                                   (short)0, acc0, false, false);
    acc1 = __builtin_amdgcn_wmma_f32_16x16x32_bf16(false, a.v, false, b1.v,
                                                   (short)0, acc1, false, false);
    __syncthreads();
  }
  #pragma unroll
  for (int nt = 0; nt < 2; ++nt) {
    v8f acc = nt ? acc1 : acc0;
    int j = n0 + wn * 32 + nt * 16 + ln;
    float bj = bo[j];
    #pragma unroll
    for (int v = 0; v < 8; ++v) {
      int m = m0 + wm * 16 + g * 8 + v;
      out[(size_t)m * HH + j] = acc[v] + bj;
    }
  }
}

// ---------------------------------------------------------------------------
extern "C" void kernel_launch(void* const* d_in, const int* in_sizes, int n_in,
                              void* d_out, int out_size, void* d_ws,
                              size_t ws_size, hipStream_t stream) {
  const float* X  = (const float*)d_in[0];
  const float* am = (const float*)d_in[1];   // attn_mask [1,NH,S,S]
  const float* km = (const float*)d_in[2];   // attention_mask [B*NH,1,S]
  const float* Wq = (const float*)d_in[3];
  const float* bq = (const float*)d_in[4];
  const float* Wk = (const float*)d_in[5];
  const float* bk = (const float*)d_in[6];
  const float* Wv = (const float*)d_in[7];
  const float* bv = (const float*)d_in[8];
  const float* Wo = (const float*)d_in[9];
  const float* bo = (const float*)d_in[10];
  float* out = (float*)d_out;

  unsigned short* ws = (unsigned short*)d_ws;
  const size_t per = (size_t)BB * NHEAD * SS * HDIM; // 4M bf16 elems
  unsigned short* qb  = ws;
  unsigned short* kb  = qb + per;
  unsigned short* vb  = kb + per;
  unsigned short* ctx = vb + per;                    // M*H bf16 elems

  dim3 g1(HH / 64, MM / 64, 3);
  qkv_gemm<<<g1, 256, 0, stream>>>(X, Wq, bq, Wk, bk, Wv, bv, qb, kb, vb);

  dim3 g2(SS / 64, BB * NHEAD, 1);
  flash_attn<<<g2, 128, 0, stream>>>(qb, kb, vb, am, km, ctx);

  dim3 g3(HH / 64, MM / 64, 1);
  out_proj<<<g3, 256, 0, stream>>>(ctx, Wo, bo, out);
}
